// CrossAttention_73366631351059
// MI455X (gfx1250) — matrix operand order repaired and checked by reading, not compiled, not run on hardware
//
#include <hip/hip_runtime.h>
#include <hip/hip_bf16.h>
#include <math.h>

// ---------------- problem constants ----------------
#define B_     8
#define SQ_    512
#define SKV_   2048
#define EMBED  512
#define HEADS  8
#define HDIM   64
#define LROW   2052   // padded LDS row stride (dwords); 2052 % 64 = 4 -> staggered banks

typedef _Float16 half_t;
typedef __attribute__((ext_vector_type(16))) _Float16 v16h;
typedef __attribute__((ext_vector_type(8)))  _Float16 v8h;
typedef __attribute__((ext_vector_type(8)))  float    v8f;

#define WMMA_F16(a, b, c) \
  __builtin_amdgcn_wmma_f32_16x16x32_f16(false, (a), false, (b), (short)0, (c), false, false)

// ---------------- fragment loaders (wave32 ISA layouts) ----------------
// A 16x32 f16: lane sub=row, hi=lane>>4; e<8 -> k=k0+8*hi+e ; e>=8 -> k=k0+16+8*hi+(e-8)
__device__ __forceinline__ v16h frag_a_f16(const half_t* __restrict__ base, size_t ld,
                                           int m0, int k0) {
  int lane = threadIdx.x & 31;
  int sub = lane & 15, hi = lane >> 4;
  const half_t* p = base + (size_t)(m0 + sub) * ld + k0 + 8 * hi;
  v8h lo = *(const v8h*)p;
  v8h hh = *(const v8h*)(p + 16);
  v16h a;
#pragma unroll
  for (int i = 0; i < 8; ++i) { a[i] = lo[i]; a[i + 8] = hh[i]; }
  return a;
}

// B 32x16 (K x N), source stored with k contiguous per n-row ("X @ W^T" layout):
// lane sub=col n, hi=lane>>4; e<8 -> k=k0+8*hi+e ; e>=8 -> k=k0+16+8*hi+(e-8)
__device__ __forceinline__ v16h frag_b_f16(const half_t* __restrict__ base, size_t ld,
                                           int n0, int k0) {
  int lane = threadIdx.x & 31;
  int sub = lane & 15, hi = lane >> 4;
  const half_t* p = base + (size_t)(n0 + sub) * ld + k0 + 8 * hi;
  v8h lo = *(const v8h*)p;
  v8h hh = *(const v8h*)(p + 16);
  v16h b;
#pragma unroll
  for (int i = 0; i < 8; ++i) { b[i] = lo[i]; b[i + 8] = hh[i]; }
  return b;
}

// ---------------- kernel 0: one-time f32 -> f16 conversion (n multiple of 8) ---
__global__ void __launch_bounds__(256)
cvt_kernel(const float* __restrict__ x, half_t* __restrict__ y, int n8) {
  int i = blockIdx.x * blockDim.x + threadIdx.x;
  if (i >= n8) return;
  const float* p = x + (size_t)i * 8;
  v8h o;
#pragma unroll
  for (int j = 0; j < 8; ++j) o[j] = (_Float16)p[j];
  *(v8h*)(y + (size_t)i * 8) = o;
}

// ---------------- kernel 1: Y(f16) = X(f16) @ W^T(f16) + bias ; optional V transpose
__global__ void __launch_bounds__(32)
proj_kernel(const half_t* __restrict__ X, const half_t* __restrict__ W,
            const float* __restrict__ bias, half_t* __restrict__ Y, int vtrans) {
  const int NT = EMBED / 32;                 // 16 n-tiles
  int m0 = (blockIdx.x / NT) * 32;
  int n0 = (blockIdx.x % NT) * 32;
  int lane = threadIdx.x & 31;
  int sub = lane & 15, hi = lane >> 4;

  v8f acc[2][2] = {};
#pragma unroll 4
  for (int k0 = 0; k0 < EMBED; k0 += 32) {
    v16h a0 = frag_a_f16(X, EMBED, m0, k0);
    v16h a1 = frag_a_f16(X, EMBED, m0 + 16, k0);
    v16h b0 = frag_b_f16(W, EMBED, n0, k0);
    v16h b1 = frag_b_f16(W, EMBED, n0 + 16, k0);
    acc[0][0] = WMMA_F16(a0, b0, acc[0][0]);
    acc[0][1] = WMMA_F16(a0, b1, acc[0][1]);
    acc[1][0] = WMMA_F16(a1, b0, acc[1][0]);
    acc[1][1] = WMMA_F16(a1, b1, acc[1][1]);
  }

#pragma unroll
  for (int i = 0; i < 2; ++i)
#pragma unroll
    for (int j = 0; j < 2; ++j) {
      int n = n0 + 16 * j + sub;
      float bn = bias[n];
      if (!vtrans) {
#pragma unroll
        for (int r = 0; r < 8; ++r) {
          int m = m0 + 16 * i + 8 * hi + r;
          Y[(size_t)m * EMBED + n] = (_Float16)(acc[i][j][r] + bn);
        }
      } else {  // (B,H,D,SKV): consecutive r -> consecutive keys -> one 16B store
        int mb = m0 + 16 * i + 8 * hi;       // key base (8 consecutive)
        int b = mb >> 11, key = mb & (SKV_ - 1);
        int h = n >> 6, d = n & 63;
        v8h o;
#pragma unroll
        for (int r = 0; r < 8; ++r) o[r] = (_Float16)(acc[i][j][r] + bn);
        *(v8h*)(Y + ((size_t)(b * HEADS + h) * HDIM + d) * SKV_ + key) = o;
      }
    }
}

// ---------------- kernel 2: fused scores + softmax + weights write + PV --------
// One workgroup = (b, h, 16 query rows). 4 waves, 128 threads.
// LDS: 16 rows x LROW f32 score strip + 16 f32 reciprocal sums.
__global__ void __launch_bounds__(128)
attn_kernel(const half_t* __restrict__ Qh, const half_t* __restrict__ Kh,
            const half_t* __restrict__ Vt, const unsigned char* __restrict__ mask,
            float* __restrict__ Wout, half_t* __restrict__ Ah) {
  extern __shared__ float sm[];
  float* sS   = sm;               // 16 * LROW
  float* sInv = sm + 16 * LROW;   // 16

  int bh = blockIdx.y;
  int b = bh >> 3, h = bh & 7;
  int m0 = blockIdx.x * 16;       // query tile base
  int wv = threadIdx.x >> 5;      // wave 0..3
  int lane = threadIdx.x & 31;
  int sub = lane & 15, hi = lane >> 4;

  const half_t* Qb = Qh + (size_t)(b * SQ_)  * EMBED + h * HDIM;
  const half_t* Kb = Kh + (size_t)(b * SKV_) * EMBED + h * HDIM;
  const half_t* Vb = Vt + (size_t)bh * HDIM * SKV_;

  // preload the 16x64 Q tile as two A fragments (reused for all key tiles)
  v16h qa0 = frag_a_f16(Qb, EMBED, m0, 0);
  v16h qa1 = frag_a_f16(Qb, EMBED, m0, 32);

  // ---- phase 1: scores for this wave's 512-key range -> LDS ----
  for (int n0 = wv * 512; n0 < wv * 512 + 512; n0 += 16) {
    v16h kb0 = frag_b_f16(Kb, EMBED, n0, 0);
    v16h kb1 = frag_b_f16(Kb, EMBED, n0, 32);
    v8f acc = {};
    acc = WMMA_F16(qa0, kb0, acc);
    acc = WMMA_F16(qa1, kb1, acc);
    int n = n0 + sub;
    bool msk = mask[b * SKV_ + n] != 0;
#pragma unroll
    for (int r = 0; r < 8; ++r) {
      int row = r + 8 * hi;
      sS[row * LROW + n] = msk ? -__builtin_inff() : acc[r] * 0.125f;
    }
  }
  __syncthreads();

  // ---- phase 2: softmax; wave wv owns rows 4*wv .. 4*wv+3 ----
  for (int rr = 0; rr < 4; ++rr) {
    int row = wv * 4 + rr;
    float* pr = sS + row * LROW;
    float mx = -__builtin_inff();
#pragma unroll 8
    for (int i = 0; i < 64; ++i) mx = fmaxf(mx, pr[lane + 32 * i]);
#pragma unroll
    for (int d = 16; d > 0; d >>= 1) mx = fmaxf(mx, __shfl_xor(mx, d, 32));
    float sum = 0.f;
#pragma unroll 8
    for (int i = 0; i < 64; ++i) {
      float e = expf(pr[lane + 32 * i] - mx);
      pr[lane + 32 * i] = e;       // keep exp in LDS; scale by inv at use
      sum += e;
    }
#pragma unroll
    for (int d = 16; d > 0; d >>= 1) sum += __shfl_xor(sum, d, 32);
    if (lane == 0) sInv[row] = 1.0f / sum;
  }
  __syncthreads();

  // ---- phase 3: single coalesced HBM write of normalized weights ----
  float* Wrow = Wout + ((size_t)bh * SQ_ + m0) * SKV_;
  for (int i = 0; i < 64; ++i) {
    int flat = threadIdx.x + 128 * i;   // float4 index over 16*2048 elements
    int row = flat >> 9;                // 512 float4 per row
    int c4 = (flat & 511) << 2;
    float inv = sInv[row];
    const float* ps = sS + row * LROW + c4;
    float4 w;
    w.x = ps[0] * inv; w.y = ps[1] * inv; w.z = ps[2] * inv; w.w = ps[3] * inv;
    *(float4*)(Wrow + (size_t)row * SKV_ + c4) = w;
  }

  // ---- phase 4: PV GEMM; wave wv owns head-dim tile n0 = 16*wv ----
  {
    int n0 = wv * 16;
    v8f acc = {};
    for (int k0 = 0; k0 < SKV_; k0 += 32) {
      const float* pa = sS + (size_t)sub * LROW + k0 + 8 * hi;
      float ainv = sInv[sub];
      v16h a;
#pragma unroll
      for (int i = 0; i < 8; ++i) {
        a[i]     = (_Float16)(pa[i]      * ainv);
        a[i + 8] = (_Float16)(pa[16 + i] * ainv);
      }
      v16h bf = frag_b_f16(Vb, SKV_, n0, k0);
      acc = WMMA_F16(a, bf, acc);
    }
#pragma unroll
    for (int r = 0; r < 8; ++r) {
      int m = m0 + 8 * hi + r;
      int n = n0 + sub;
      Ah[(size_t)(b * SQ_ + m) * EMBED + h * HDIM + n] = (_Float16)acc[r];
    }
  }
}

// ---------------- kernel 3: out = attn(f16) @ Wo^T(f16) + bo -> f32 ------------
__global__ void __launch_bounds__(32)
oproj_kernel(const half_t* __restrict__ X, const half_t* __restrict__ W,
             const float* __restrict__ bias, float* __restrict__ Y) {
  const int NT = EMBED / 32;
  int m0 = (blockIdx.x / NT) * 32;
  int n0 = (blockIdx.x % NT) * 32;
  int lane = threadIdx.x & 31;
  int sub = lane & 15, hi = lane >> 4;

  v8f acc[2][2] = {};
#pragma unroll 4
  for (int k0 = 0; k0 < EMBED; k0 += 32) {
    v16h a0 = frag_a_f16(X, EMBED, m0, k0);
    v16h a1 = frag_a_f16(X, EMBED, m0 + 16, k0);
    v16h b0 = frag_b_f16(W, EMBED, n0, k0);
    v16h b1 = frag_b_f16(W, EMBED, n0 + 16, k0);
    acc[0][0] = WMMA_F16(a0, b0, acc[0][0]);
    acc[0][1] = WMMA_F16(a0, b1, acc[0][1]);
    acc[1][0] = WMMA_F16(a1, b0, acc[1][0]);
    acc[1][1] = WMMA_F16(a1, b1, acc[1][1]);
  }

#pragma unroll
  for (int i = 0; i < 2; ++i)
#pragma unroll
    for (int j = 0; j < 2; ++j) {
      int n = n0 + 16 * j + sub;
      float bn = bias[n];
#pragma unroll
      for (int r = 0; r < 8; ++r) {
        int m = m0 + 16 * i + 8 * hi + r;
        Y[(size_t)m * EMBED + n] = acc[i][j][r] + bn;
      }
    }
}

// ---------------- host launch ----------------
extern "C" void kernel_launch(void* const* d_in, const int* in_sizes, int n_in,
                              void* d_out, int out_size, void* d_ws, size_t ws_size,
                              hipStream_t stream) {
  (void)in_sizes; (void)n_in; (void)out_size; (void)ws_size;
  const float* query = (const float*)d_in[0];
  const float* kv    = (const float*)d_in[1];
  const unsigned char* mask = (const unsigned char*)d_in[2];
  const float* Wq = (const float*)d_in[3];
  const float* bq = (const float*)d_in[4];
  const float* Wk = (const float*)d_in[5];
  const float* bk = (const float*)d_in[6];
  const float* Wv = (const float*)d_in[7];
  const float* bv = (const float*)d_in[8];
  const float* Wo = (const float*)d_in[9];
  const float* bo = (const float*)d_in[10];

  float* out = (float*)d_out;                                  // (B,SQ,EMBED)
  float* wgt = out + (size_t)B_ * SQ_ * EMBED;                 // (B,H,SQ,SKV)

  const size_t NQ = (size_t)B_ * SQ_ * EMBED;    // 2,097,152
  const size_t NK = (size_t)B_ * SKV_ * EMBED;   // 8,388,608
  const size_t NW = (size_t)EMBED * EMBED;       // 262,144

  half_t* p   = (half_t*)d_ws;
  half_t* Qin = p;            p += NQ;   // f16 query
  half_t* KVi = p;            p += NK;   // f16 key_value
  half_t* Wqh = p;            p += NW;
  half_t* Wkh = p;            p += NW;
  half_t* Wvh = p;            p += NW;
  half_t* Woh = p;            p += NW;
  half_t* Qh  = p;            p += NQ;   // projected Q  (B*SQ, EMBED)
  half_t* Kh  = p;            p += NK;   // projected K  (B*SKV, EMBED)
  half_t* Vt  = p;            p += NK;   // projected V  (B,H,D,SKV)
  half_t* Ah  = p;            p += NQ;   // attention output (B*SQ, EMBED)

  // one-time f32 -> f16 conversions
  cvt_kernel<<<dim3((int)(NQ / 8 / 256)), 256, 0, stream>>>(query, Qin, (int)(NQ / 8));
  cvt_kernel<<<dim3((int)(NK / 8 / 256)), 256, 0, stream>>>(kv, KVi, (int)(NK / 8));
  cvt_kernel<<<dim3((int)(NW / 8 / 256)), 256, 0, stream>>>(Wq, Wqh, (int)(NW / 8));
  cvt_kernel<<<dim3((int)(NW / 8 / 256)), 256, 0, stream>>>(Wk, Wkh, (int)(NW / 8));
  cvt_kernel<<<dim3((int)(NW / 8 / 256)), 256, 0, stream>>>(Wv, Wvh, (int)(NW / 8));
  cvt_kernel<<<dim3((int)(NW / 8 / 256)), 256, 0, stream>>>(Wo, Woh, (int)(NW / 8));

  // projections (WMMA f16, f32 accumulate)
  proj_kernel<<<dim3((B_ * SQ_ / 32) * (EMBED / 32)), 32, 0, stream>>>(Qin, Wqh, bq, Qh, 0);
  proj_kernel<<<dim3((B_ * SKV_ / 32) * (EMBED / 32)), 32, 0, stream>>>(KVi, Wkh, bk, Kh, 0);
  proj_kernel<<<dim3((B_ * SKV_ / 32) * (EMBED / 32)), 32, 0, stream>>>(KVi, Wvh, bv, Vt, 1);

  // fused attention: scores + softmax + weights write + PV (LDS-resident strip)
  size_t smem = (size_t)(16 * LROW + 16) * sizeof(float);      // ~128.4 KB
  attn_kernel<<<dim3(SQ_ / 16, B_ * HEADS), 128, smem, stream>>>(Qh, Kh, Vt, mask, wgt, Ah);

  // output projection -> f32
  oproj_kernel<<<dim3((B_ * SQ_ / 32) * (EMBED / 32)), 32, 0, stream>>>(Ah, Woh, bo, out);
}
